// ResidualVQVAE_30666066493668
// MI455X (gfx1250) — matrix operand order repaired and checked by reading, not compiled, not run on hardware
//
#include <hip/hip_runtime.h>
#include <hip/hip_bf16.h>
#include <math.h>

// ---------------- model constants ----------------
#define BATCH   262144
#define IN_DIM  54
#define HID     256
#define EXP     512
#define LAT     32
#define NBLK    4
#define NCODES  512

typedef _Float16 f16;
typedef f16  v16h __attribute__((ext_vector_type(16)));
typedef f16  v8h  __attribute__((ext_vector_type(8)));
typedef float v8f __attribute__((ext_vector_type(8)));

// ---------------- workspace layout (in f16 elements for fragments) ----------
// Fragments: each WMMA B-fragment = 512 halves (32 lanes x 16 halves), 1KB.
#define OFF_ENC_PROJ 0ull           // KC=2  NT=16 -> 32 frags
#define OFF_ENC_W1   16384ull       // 4 x (KC=8  NT=32 -> 256 frags)
#define OFF_ENC_W2   540672ull      // 4 x (KC=16 NT=16 -> 256 frags)
#define OFF_ENC_LAT  1064960ull     // KC=8  NT=2  -> 16 frags
#define OFF_CBT      1073152ull     // KC=1  NT=32 -> 32 frags (codebook^T)
#define OFF_DEC_LAT  1089536ull     // KC=1  NT=16 -> 16 frags
#define OFF_DEC_W1   1097728ull
#define OFF_DEC_W2   1622016ull
#define OFF_DEC_PROJ 2146304ull     // KC=8  NT=4  -> 32 frags
#define FRAG_HALVES  2162688ull     // total halves
#define WS_CNORM_OFF (FRAG_HALVES*2ull)           // 512 floats
#define WS_PART_OFF  (WS_CNORM_OFF + 2048ull)     // 16384 floats (per-block commit partials)

#define NTILES (BATCH/16)   // 16384

// ---------------- helpers ----------------
__device__ __forceinline__ v8f zero8() {
  v8f z = {0.f,0.f,0.f,0.f,0.f,0.f,0.f,0.f};
  return z;
}

__device__ __forceinline__ float gelu_erf(float x) {
  return 0.5f * x * (1.0f + erff(x * 0.70710678118654752f));
}

// Assemble A-fragment (16x32 f16) for one K-chunk from a row-major f16 LDS
// buffer. A layout: lanes 0-15 row M=lane, halves 0..15 = K 0..15;
// lanes 16-31 get K 16..31 (matches pack_w B layout).
__device__ __forceinline__ v16h load_a_frag(const f16* buf, int pitch, int kc, int lane) {
  int row = lane & 15;
  int kb  = kc * 32 + ((lane & 16) ? 16 : 0);
  const f16* p = buf + row * pitch + kb;
  v8h lo = *(const v8h*)(p);
  v8h hi = *(const v8h*)(p + 8);
  return __builtin_shufflevector(lo, hi, 0,1,2,3,4,5,6,7,8,9,10,11,12,13,14,15);
}

__device__ __forceinline__ v8f wmma_f16(v16h a, v16h b, v8f c) {
  return __builtin_amdgcn_wmma_f32_16x16x32_f16(false, a, false, b, (short)0, c, false, false);
}

// ---------------- pack kernel: fp32 weight -> f16 B-fragments ----------------
// B-fragment layout (K=32 x N=16 f16): lane l, half h:
//   N = tileN*16 + (l & 15), K = kc*32 + h + ((l & 16) ? 16 : 0)
__global__ void pack_w(const float* __restrict__ src, f16* __restrict__ dst,
                       int Korig, int Norig, int ld, int NT, int transposeSrc) {
  int f    = blockIdx.x;
  int kc   = f / NT;
  int nt   = f % NT;
  int lane = threadIdx.x;
  int n  = nt * 16 + (lane & 15);
  int kb = kc * 32 + ((lane & 16) ? 16 : 0);
#pragma unroll
  for (int h = 0; h < 16; ++h) {
    int k = kb + h;
    float v = 0.0f;
    if (k < Korig && n < Norig)
      v = transposeSrc ? src[(size_t)n * ld + k] : src[(size_t)k * ld + n];
    dst[(size_t)f * 512 + lane * 16 + h] = (f16)v;
  }
}

// codebook row norms
__global__ void cnorm_kernel(const float* __restrict__ cb, float* __restrict__ cn) {
  int n = blockIdx.x * blockDim.x + threadIdx.x;
  if (n < NCODES) {
    float s = 0.f;
#pragma unroll
    for (int k = 0; k < LAT; ++k) { float v = cb[n * LAT + k]; s += v * v; }
    cn[n] = s;
  }
}

// deterministic commit-loss reduction (fixed order)
__global__ void commit_reduce(const float* __restrict__ part, float* __restrict__ out) {
  __shared__ float sh[256];
  float s = 0.f;
  for (int i = threadIdx.x; i < NTILES; i += 256) s += part[i];
  sh[threadIdx.x] = s;
  __syncthreads();
  if (threadIdx.x == 0) {
    float t = 0.f;
    for (int i = 0; i < 256; ++i) t += sh[i];
    *out = t / ((float)BATCH * (float)LAT);
  }
}

// ---------------- shared memory plan ----------------
// Aliased union: score buffer reuses h/y/t after encoder is finished.
union SMemU {
  struct {
    float hbuf[16 * 264];   // 16896 B  residual stream (fp32)
    f16   ybuf[16 * 264];   //  8448 B  LN output (f16, A-source)
    f16   tbuf[16 * 520];   // 16640 B  expansion activations (f16); also x input stage
  } a;
  float sbuf[16 * 512];     // 32768 B  VQ scores (aliases the above)
};

// ---------------- fused VQ-VAE kernel ----------------
__device__ __forceinline__ void layernorm16(const float* hbuf, f16* ybuf,
                                            const float* g, const float* b,
                                            float* red_a, float* red_b,
                                            float* mu_s, float* rs_s, int tid) {
  int r = tid & 15, s = tid >> 4;
  float sm = 0.f, sq = 0.f;
#pragma unroll
  for (int j = 0; j < 16; ++j) {
    float v = hbuf[r * 264 + s * 16 + j];
    sm += v; sq += v * v;
  }
  red_a[tid] = sm; red_b[tid] = sq;
  __syncthreads();
  if (tid < 16) {
    float S = 0.f, Q = 0.f;
    for (int ss = 0; ss < 16; ++ss) { S += red_a[ss * 16 + tid]; Q += red_b[ss * 16 + tid]; }
    float mu  = S * (1.0f / 256.0f);
    float var = Q * (1.0f / 256.0f) - mu * mu;
    mu_s[tid] = mu;
    rs_s[tid] = rsqrtf(var + 1e-5f);
  }
  __syncthreads();
  float mu = mu_s[r], rs = rs_s[r];
#pragma unroll
  for (int j = 0; j < 16; ++j) {
    int c = s * 16 + j;
    float v = (hbuf[r * 264 + c] - mu) * rs * g[c] + b[c];
    ybuf[r * 264 + c] = (f16)v;
  }
  __syncthreads();
}

// K-chunk-outer matmuls: one A-fragment load feeds all N-tiles (independent
// accumulators -> back-to-back independent v_wmma, minimal LDS A traffic).
__device__ __forceinline__ void mlp_block(float* hbuf, const f16* ybuf, f16* tbuf,
                                          const f16* __restrict__ W1, const f16* __restrict__ W2,
                                          const float* __restrict__ b1, const float* __restrict__ b2,
                                          int lane, int wave) {
  const int col16 = lane & 15;
  const int mb    = (lane & 16) ? 8 : 0;
  // y(16x256) @ w1(256x512) + b1 -> gelu -> tbuf (f16)
  {
    v8f acc[4];
#pragma unroll
    for (int t = 0; t < 4; ++t) acc[t] = zero8();
#pragma unroll
    for (int kc = 0; kc < 8; ++kc) {
      v16h a = load_a_frag(ybuf, 264, kc, lane);
#pragma unroll
      for (int t = 0; t < 4; ++t) {
        int nt = wave + t * 8;
        v16h b = *(const v16h*)(W1 + (size_t)(kc * 32 + nt) * 512 + lane * 16);
        acc[t] = wmma_f16(a, b, acc[t]);
      }
    }
#pragma unroll
    for (int t = 0; t < 4; ++t) {
      int col = (wave + t * 8) * 16 + col16;
      float bias = b1[col];
#pragma unroll
      for (int r = 0; r < 8; ++r)
        tbuf[(mb + r) * 520 + col] = (f16)gelu_erf(acc[t][r] + bias);
    }
  }
  __syncthreads();
  // t(16x512) @ w2(512x256) + b2 -> h += (fp32 residual)
  {
    v8f acc[2];
    acc[0] = zero8(); acc[1] = zero8();
#pragma unroll
    for (int kc = 0; kc < 16; ++kc) {
      v16h a = load_a_frag(tbuf, 520, kc, lane);
#pragma unroll
      for (int t = 0; t < 2; ++t) {
        int nt = wave + t * 8;
        v16h b = *(const v16h*)(W2 + (size_t)(kc * 16 + nt) * 512 + lane * 16);
        acc[t] = wmma_f16(a, b, acc[t]);
      }
    }
#pragma unroll
    for (int t = 0; t < 2; ++t) {
      int col = (wave + t * 8) * 16 + col16;
      float bias = b2[col];
#pragma unroll
      for (int r = 0; r < 8; ++r)
        hbuf[(mb + r) * 264 + col] += acc[t][r] + bias;
    }
  }
  __syncthreads();
}

__global__ __launch_bounds__(256) void vqvae_fused(
    const float* __restrict__ x,
    const float* __restrict__ enc_proj_b,
    const float* __restrict__ enc_ln_g, const float* __restrict__ enc_ln_b,
    const float* __restrict__ enc_b1,   const float* __restrict__ enc_b2,
    const float* __restrict__ enc_norm_g, const float* __restrict__ enc_norm_b,
    const float* __restrict__ enc_lat_b,
    const float* __restrict__ codebook,
    const float* __restrict__ dec_lat_b,
    const float* __restrict__ dec_ln_g, const float* __restrict__ dec_ln_b,
    const float* __restrict__ dec_b1,   const float* __restrict__ dec_b2,
    const float* __restrict__ dec_norm_g, const float* __restrict__ dec_norm_b,
    const float* __restrict__ dec_proj_b,
    const f16* __restrict__ wsf,
    const float* __restrict__ cnorm,
    float* __restrict__ out_x, float* __restrict__ out_zq,
    int* __restrict__ out_idx, float* __restrict__ commit_part) {

  __shared__ SMemU u;
  __shared__ f16   zbuf16[16 * 40];
  __shared__ f16   zq16s [16 * 40];
  __shared__ float zf32  [16 * 32];
  __shared__ float red_a[256], red_b[256];
  __shared__ int   red_i[256];
  __shared__ float mu_s[16], rs_s[16];
  __shared__ int   idx_s[16];

  const int tid   = threadIdx.x;
  const int lane  = tid & 31;
  const int wave  = tid >> 5;
  const int r16   = tid & 15;
  const int seg   = tid >> 4;
  const int col16 = lane & 15;
  const int mb    = (lane & 16) ? 8 : 0;
  const int grow0 = blockIdx.x * 16;

  // ---------- load x tile into f16 A-source (pad 54 -> 64), alias tbuf ----------
#pragma unroll
  for (int j = 0; j < 4; ++j) {
    int c = seg * 4 + j;
    float v = (c < IN_DIM) ? x[(size_t)(grow0 + r16) * IN_DIM + c] : 0.0f;
    u.a.tbuf[r16 * 72 + c] = (f16)v;
  }
  __syncthreads();

  // ---------- encoder projection: x(16x64) @ Wp(64x256) + b ----------
  {
    const f16* Wp = wsf + OFF_ENC_PROJ;
    v8f acc[2];
    acc[0] = zero8(); acc[1] = zero8();
#pragma unroll
    for (int kc = 0; kc < 2; ++kc) {
      v16h a = load_a_frag(u.a.tbuf, 72, kc, lane);
#pragma unroll
      for (int t = 0; t < 2; ++t) {
        int nt = wave + t * 8;
        v16h b = *(const v16h*)(Wp + (size_t)(kc * 16 + nt) * 512 + lane * 16);
        acc[t] = wmma_f16(a, b, acc[t]);
      }
    }
#pragma unroll
    for (int t = 0; t < 2; ++t) {
      int col = (wave + t * 8) * 16 + col16;
      float bias = enc_proj_b[col];
#pragma unroll
      for (int r = 0; r < 8; ++r)
        u.a.hbuf[(mb + r) * 264 + col] = acc[t][r] + bias;
    }
  }
  __syncthreads();

  // ---------- encoder blocks ----------
  for (int i = 0; i < NBLK; ++i) {
    layernorm16(u.a.hbuf, u.a.ybuf, enc_ln_g + i * HID, enc_ln_b + i * HID,
                red_a, red_b, mu_s, rs_s, tid);
    mlp_block(u.a.hbuf, u.a.ybuf, u.a.tbuf,
              wsf + OFF_ENC_W1 + (size_t)i * 131072,
              wsf + OFF_ENC_W2 + (size_t)i * 131072,
              enc_b1 + i * EXP, enc_b2 + i * HID, lane, wave);
  }

  // ---------- final encoder LN ----------
  layernorm16(u.a.hbuf, u.a.ybuf, enc_norm_g, enc_norm_b, red_a, red_b, mu_s, rs_s, tid);

  // ---------- latent projection: y(16x256) @ Wl(256x32) ----------
  if (wave < 2) {
    int nt = wave;
    v8f acc = zero8();
#pragma unroll
    for (int kc = 0; kc < 8; ++kc) {
      v16h a = load_a_frag(u.a.ybuf, 264, kc, lane);
      v16h b = *(const v16h*)(wsf + OFF_ENC_LAT + (size_t)(kc * 2 + nt) * 512 + lane * 16);
      acc = wmma_f16(a, b, acc);
    }
    int col = nt * 16 + col16;
    float bias = enc_lat_b[col];
#pragma unroll
    for (int r = 0; r < 8; ++r) {
      float v = acc[r] + bias;
      zf32  [(mb + r) * 32 + col] = v;
      zbuf16[(mb + r) * 40 + col] = (f16)v;
    }
  }
  __syncthreads();

  // ---------- VQ scores: cnorm[n] - 2 * (z @ cb^T) ----------
  {
    v16h a = load_a_frag(zbuf16, 40, 0, lane);   // shared by all 4 tiles
#pragma unroll
    for (int t = 0; t < 4; ++t) {
      int nt = wave + t * 8;
      v8f acc = zero8();
      v16h b = *(const v16h*)(wsf + OFF_CBT + (size_t)nt * 512 + lane * 16);
      acc = wmma_f16(a, b, acc);
      int n = nt * 16 + col16;
      float cn = cnorm[n];
#pragma unroll
      for (int r = 0; r < 8; ++r)
        u.sbuf[(mb + r) * 512 + n] = cn - 2.0f * acc[r];
    }
  }
  __syncthreads();

  // ---------- argmin over 512 codes (first-occurrence tie-break) ----------
  {
    float best = 3.4e38f; int bi = 0;
#pragma unroll 4
    for (int j = 0; j < 32; ++j) {
      int n = seg * 32 + j;
      float v = u.sbuf[r16 * 512 + n];
      if (v < best) { best = v; bi = n; }
    }
    red_a[tid] = best; red_i[tid] = bi;
  }
  __syncthreads();
  if (tid < 16) {
    float best = 3.4e38f; int bi = 0;
    for (int s = 0; s < 16; ++s) {
      float v = red_a[s * 16 + tid];
      if (v < best) { best = v; bi = red_i[s * 16 + tid]; }
    }
    idx_s[tid] = bi;
    out_idx[grow0 + tid] = bi;
  }
  __syncthreads();

  // ---------- z_q lookup, outputs, commit partial ----------
  {
    int ci = idx_s[r16];
    float csum = 0.f;
#pragma unroll
    for (int j = 0; j < 2; ++j) {
      int k = seg * 2 + j;
      float q = codebook[(size_t)ci * LAT + k];
      out_zq[(size_t)(grow0 + r16) * LAT + k] = q;   // z_q_st == z_q numerically
      zq16s[r16 * 40 + k] = (f16)q;
      float d = zf32[r16 * 32 + k] - q;
      csum += d * d;
    }
    red_b[tid] = csum;
  }
  __syncthreads();
  if (tid == 0) {
    float tot = 0.f;
    for (int i = 0; i < 256; ++i) tot += red_b[i];
    commit_part[blockIdx.x] = tot;
  }
  __syncthreads();

  // ---------- decoder latent: zq(16x32) @ Wd(32x256) ----------
  {
    const f16* Wd = wsf + OFF_DEC_LAT;
    v16h a = load_a_frag(zq16s, 40, 0, lane);    // shared by both tiles
#pragma unroll
    for (int t = 0; t < 2; ++t) {
      int nt = wave + t * 8;
      v8f acc = zero8();
      v16h b = *(const v16h*)(Wd + (size_t)nt * 512 + lane * 16);
      acc = wmma_f16(a, b, acc);
      int col = nt * 16 + col16;
      float bias = dec_lat_b[col];
#pragma unroll
      for (int r = 0; r < 8; ++r)
        u.a.hbuf[(mb + r) * 264 + col] = acc[r] + bias;
    }
  }
  __syncthreads();

  // ---------- decoder blocks ----------
  for (int i = 0; i < NBLK; ++i) {
    layernorm16(u.a.hbuf, u.a.ybuf, dec_ln_g + i * HID, dec_ln_b + i * HID,
                red_a, red_b, mu_s, rs_s, tid);
    mlp_block(u.a.hbuf, u.a.ybuf, u.a.tbuf,
              wsf + OFF_DEC_W1 + (size_t)i * 131072,
              wsf + OFF_DEC_W2 + (size_t)i * 131072,
              dec_b1 + i * EXP, dec_b2 + i * HID, lane, wave);
  }

  // ---------- final decoder LN ----------
  layernorm16(u.a.hbuf, u.a.ybuf, dec_norm_g, dec_norm_b, red_a, red_b, mu_s, rs_s, tid);

  // ---------- output projection: y(16x256) @ Wo(256x64pad) -> cols 0..53 ----------
  if (wave < 4) {
    int nt = wave;
    v8f acc = zero8();
#pragma unroll
    for (int kc = 0; kc < 8; ++kc) {
      v16h a = load_a_frag(u.a.ybuf, 264, kc, lane);
      v16h b = *(const v16h*)(wsf + OFF_DEC_PROJ + (size_t)(kc * 4 + nt) * 512 + lane * 16);
      acc = wmma_f16(a, b, acc);
    }
    int col = nt * 16 + col16;
    if (col < IN_DIM) {
      float bias = dec_proj_b[col];
#pragma unroll
      for (int r = 0; r < 8; ++r)
        out_x[(size_t)(grow0 + mb + r) * IN_DIM + col] = acc[r] + bias;
    }
  }
}

// ---------------- host launch ----------------
extern "C" void kernel_launch(void* const* d_in, const int* in_sizes, int n_in,
                              void* d_out, int out_size, void* d_ws, size_t ws_size,
                              hipStream_t stream) {
  const float* x          = (const float*)d_in[0];
  const float* enc_proj_w = (const float*)d_in[1];
  const float* enc_proj_b = (const float*)d_in[2];
  const float* enc_ln_g   = (const float*)d_in[3];
  const float* enc_ln_b   = (const float*)d_in[4];
  const float* enc_w1     = (const float*)d_in[5];
  const float* enc_b1     = (const float*)d_in[6];
  const float* enc_w2     = (const float*)d_in[7];
  const float* enc_b2     = (const float*)d_in[8];
  const float* enc_norm_g = (const float*)d_in[9];
  const float* enc_norm_b = (const float*)d_in[10];
  const float* enc_lat_w  = (const float*)d_in[11];
  const float* enc_lat_b  = (const float*)d_in[12];
  const float* codebook   = (const float*)d_in[13];
  const float* dec_lat_w  = (const float*)d_in[14];
  const float* dec_lat_b  = (const float*)d_in[15];
  const float* dec_ln_g   = (const float*)d_in[16];
  const float* dec_ln_b   = (const float*)d_in[17];
  const float* dec_w1     = (const float*)d_in[18];
  const float* dec_b1     = (const float*)d_in[19];
  const float* dec_w2     = (const float*)d_in[20];
  const float* dec_b2     = (const float*)d_in[21];
  const float* dec_norm_g = (const float*)d_in[22];
  const float* dec_norm_b = (const float*)d_in[23];
  const float* dec_proj_w = (const float*)d_in[24];
  const float* dec_proj_b = (const float*)d_in[25];

  f16*   wsf  = (f16*)d_ws;
  float* cn   = (float*)((char*)d_ws + WS_CNORM_OFF);
  float* part = (float*)((char*)d_ws + WS_PART_OFF);

  float* out     = (float*)d_out;
  float* out_x   = out;                                  // B x 54
  float* out_zq  = out + (size_t)BATCH * IN_DIM;         // B x 32
  int*   out_idx = (int*)(out + (size_t)BATCH * 86);     // B (int32 bits)
  float* out_cl  = out + (size_t)BATCH * 87;             // scalar

  // ---- pack weights to f16 WMMA B-fragments ----
  pack_w<<<32, 32, 0, stream>>>(enc_proj_w, wsf + OFF_ENC_PROJ, IN_DIM, HID, HID, 16, 0);
  for (int i = 0; i < NBLK; ++i) {
    pack_w<<<256, 32, 0, stream>>>(enc_w1 + (size_t)i * HID * EXP,
                                   wsf + OFF_ENC_W1 + (size_t)i * 131072, HID, EXP, EXP, 32, 0);
    pack_w<<<256, 32, 0, stream>>>(enc_w2 + (size_t)i * EXP * HID,
                                   wsf + OFF_ENC_W2 + (size_t)i * 131072, EXP, HID, HID, 16, 0);
    pack_w<<<256, 32, 0, stream>>>(dec_w1 + (size_t)i * HID * EXP,
                                   wsf + OFF_DEC_W1 + (size_t)i * 131072, HID, EXP, EXP, 32, 0);
    pack_w<<<256, 32, 0, stream>>>(dec_w2 + (size_t)i * EXP * HID,
                                   wsf + OFF_DEC_W2 + (size_t)i * 131072, EXP, HID, HID, 16, 0);
  }
  pack_w<<<16, 32, 0, stream>>>(enc_lat_w, wsf + OFF_ENC_LAT, HID, LAT, LAT, 2, 0);
  pack_w<<<32, 32, 0, stream>>>(codebook,  wsf + OFF_CBT,     LAT, NCODES, LAT, 32, 1);
  pack_w<<<16, 32, 0, stream>>>(dec_lat_w, wsf + OFF_DEC_LAT, LAT, HID, HID, 16, 0);
  pack_w<<<32, 32, 0, stream>>>(dec_proj_w, wsf + OFF_DEC_PROJ, HID, IN_DIM, IN_DIM, 4, 0);
  cnorm_kernel<<<2, 256, 0, stream>>>(codebook, cn);

  // ---- fused forward ----
  vqvae_fused<<<NTILES, 256, 0, stream>>>(
      x, enc_proj_b, enc_ln_g, enc_ln_b, enc_b1, enc_b2, enc_norm_g, enc_norm_b,
      enc_lat_b, codebook, dec_lat_b, dec_ln_g, dec_ln_b, dec_b1, dec_b2,
      dec_norm_g, dec_norm_b, dec_proj_b, wsf, cn, out_x, out_zq, out_idx, part);

  // ---- deterministic commit-loss reduction ----
  commit_reduce<<<1, 256, 0, stream>>>(part, out_cl);
}
